// MultiHeadAttention_34583076667449
// MI455X (gfx1250) — matrix operand order repaired and checked by reading, not compiled, not run on hardware
//
#include <hip/hip_runtime.h>
#include <hip/hip_bf16.h>
#include <math.h>

// ---------------- problem constants ----------------
#define EMBED  1024
#define NHEADS 16
#define HDIM   64
#define BATCH  4
#define SEQ    2048
#define MTOK   (BATCH * SEQ)   // 8192 tokens

typedef __bf16 bf16_t;
typedef __attribute__((ext_vector_type(16))) __bf16 v16bf;  // 8 VGPRs
typedef __attribute__((ext_vector_type(8)))  __bf16 v8bf;   // 4 VGPRs (16B)
typedef __attribute__((ext_vector_type(8)))  float  v8f;    // 8 VGPRs
typedef __attribute__((ext_vector_type(4)))  unsigned int v4u;
typedef __attribute__((ext_vector_type(4)))  int v4i;
typedef __attribute__((ext_vector_type(8)))  int v8i;

// ---------------- TDM availability ----------------
#define TDM_AVAILABLE 0
#if defined(__has_builtin)
#if __has_builtin(__builtin_amdgcn_tensor_load_to_lds)
#undef TDM_AVAILABLE
#define TDM_AVAILABLE 1
#endif
#endif

// ---------------- WMMA fragment helpers ----------------
// A-fragment (16x32 bf16, row-major source):
//  lane L: row M = L&15, g = L>>4
//  v0..v3 hold K = k0+8g .. +7 ; v4..v7 hold K = k0+16+8g .. +7 (2x16B loads)
__device__ inline v16bf load_a_frag(const bf16_t* __restrict__ A, int lda,
                                    int m0, int k0, int lane) {
  int r = lane & 15;
  int g = lane >> 4;
  const bf16_t* p = A + (size_t)(m0 + r) * lda + (k0 + 8 * g);
  union { v16bf v; v8bf h[2]; } u;
  u.h[0] = *(const v8bf*)p;
  u.h[1] = *(const v8bf*)(p + 16);
  return u.v;
}

// B-fragment (32x16 bf16) with B[k][n] = Wn[n][k] (X @ Wn^T pattern, Wn row-major
// N x K): lane L: column n = L&15, g = L>>4, K = k0+8g..+7 and k0+16+8g..+7 (2x16B loads).
__device__ inline v16bf load_b_fragT(const bf16_t* Wn, int ldw,
                                     int n0, int k0, int lane) {
  int c = lane & 15;
  int g = lane >> 4;
  const bf16_t* p = Wn + (size_t)(n0 + c) * ldw + (k0 + 8 * g);
  union { v16bf v; v8bf h[2]; } u;
  u.h[0] = *(const v8bf*)p;
  u.h[1] = *(const v8bf*)(p + 16);
  return u.v;
}

__device__ inline v8f wmma_bf16(v16bf a, v16bf b, v8f c) {
  return __builtin_amdgcn_wmma_f32_16x16x32_bf16(
      false, a, false, b, (short)0, c, false, false);
}

// ---------------- TDM 2D tile load (ISA 8.3/8.4 D# layout) ----------------
__device__ inline unsigned int lds_offset(const void* p) {
  return (unsigned int)(unsigned long long)p;   // low 32b of flat LDS address
}

#if TDM_AVAILABLE
__device__ inline void tdm_load_2d(unsigned int lds_off,
                                   unsigned long long gaddr,
                                   unsigned int tile_d0, unsigned int tile_d1,
                                   unsigned long long stride0,
                                   unsigned int tdim0, unsigned int tdim1) {
  // D# group 0: count=1 | lds_addr | global_addr[56:0] | type=2
  v4u g0;
  g0[0] = 1u;
  g0[1] = lds_off;
  g0[2] = (unsigned int)gaddr;
  g0[3] = (unsigned int)((gaddr >> 32) & 0x1ffffffull) | 0x80000000u;
  // D# group 1: data_size=2B, 2D tile/tensor dims, dim0 stride (elements)
  unsigned int w[8] = {0, 0, 0, 0, 0, 0, 0, 0};
  w[0] = 1u << 16;                                            // data_size=2B
  w[1] = (tdim0 & 0xffffu) << 16;                             // tensor_dim0[79:48]
  w[2] = ((tdim0 >> 16) & 0xffffu) | ((tdim1 & 0xffffu) << 16);
  w[3] = ((tdim1 >> 16) & 0xffffu) | ((tile_d0 & 0xffffu) << 16); // tile_dim0
  w[4] = (tile_d1 & 0xffffu);                                 // tile_dim1 (tile_dim2=0)
  w[5] = (unsigned int)stride0;                               // tensor_dim0_stride
  w[6] = (unsigned int)((stride0 >> 32) & 0xffffull);
  w[7] = 0;
  v8i g1;
  #pragma unroll
  for (int i = 0; i < 8; ++i) g1[i] = (int)w[i];
  v4i gz = {0, 0, 0, 0};
#if __clang_major__ >= 23
  v8i gz8 = {0, 0, 0, 0, 0, 0, 0, 0};
  __builtin_amdgcn_tensor_load_to_lds(g0, g1, gz, gz, gz8, 0);
#else
  __builtin_amdgcn_tensor_load_to_lds(g0, g1, gz, gz, 0);
#endif
}

__device__ inline void tensor_wait() {
#if __has_builtin(__builtin_amdgcn_s_wait_tensorcnt)
  __builtin_amdgcn_s_wait_tensorcnt((short)0);
#else
  asm volatile("s_wait_tensorcnt 0" ::: "memory");
#endif
}
#endif  // TDM_AVAILABLE

// Fill shared K tile (32 keys x 64 dims, row-major) and V tile (64 dims x 32
// keys, from transposed V) for key block kb. TDM path: wave 0 issues 2 DMAs.
__device__ inline void fill_tiles(bf16_t* Kd, bf16_t* Vd,
                                  const bf16_t* Kg, const bf16_t* Vg,
                                  int kb, int tid) {
#if TDM_AVAILABLE
  if (tid < 32) {
    // K tile: 32 contiguous rows of 64 elems (stride 64)
    tdm_load_2d(lds_offset(Kd),
                (unsigned long long)(size_t)(Kg + (size_t)kb * HDIM),
                /*tile_d0=*/HDIM, /*tile_d1=*/32, /*stride0=*/HDIM,
                /*tdim0=*/HDIM, /*tdim1=*/SEQ - kb);
    // V tile: 64 rows (dims) x 32 keys, row stride SEQ
    tdm_load_2d(lds_offset(Vd),
                (unsigned long long)(size_t)(Vg + kb),
                /*tile_d0=*/32, /*tile_d1=*/HDIM, /*stride0=*/SEQ,
                /*tdim0=*/SEQ - kb, /*tdim1=*/HDIM);
  }
#else
  // cooperative fallback: 256 threads x 16B
  *(v8bf*)(Kd + tid * 8) = *(const v8bf*)(Kg + (size_t)kb * HDIM + tid * 8);
  int vr = tid >> 2, vs = tid & 3;
  *(v8bf*)(Vd + vr * 32 + vs * 8) =
      *(const v8bf*)(Vg + (size_t)vr * SEQ + kb + vs * 8);
#endif
}

__device__ inline void fill_wait(int tid) {
#if TDM_AVAILABLE
  if (tid < 32) tensor_wait();
#else
  (void)tid;
#endif
}

// ---------------- f32 -> bf16 conversion ----------------
__global__ void cvt_f32_bf16(const float* __restrict__ in,
                             bf16_t* __restrict__ out, int n) {
  int i = blockIdx.x * blockDim.x + threadIdx.x;
  int stride = gridDim.x * blockDim.x;
  for (; i < n; i += stride) out[i] = (bf16_t)in[i];
}

// ---------------- GEMM: out = (X @ W^T + bias) * oscale ----------------
// 32x32 output block per wave: 4 WMMAs per 4 operand loads per K-step.
__device__ inline void store_tile(v8f acc, int m0, int n0, int lane,
                                  const float* __restrict__ bias,
                                  bf16_t* __restrict__ obf,
                                  float* __restrict__ of32,
                                  int N, int mode, float oscale) {
  int c = lane & 15, g = lane >> 4;
  int n = n0 + c;
  float bval = bias ? bias[n] : 0.0f;
  #pragma unroll
  for (int j = 0; j < 8; ++j) {
    int m = m0 + j + 8 * g;                 // C layout: VGPR j -> row j + 8*g
    float val = (acc[j] + bval) * oscale;
    if (mode == 2) {
      of32[(size_t)m * N + n] = val;
    } else {
      int b_ = m >> 11, s = m & (SEQ - 1);  // m = b*SEQ + s
      int h  = n >> 6,  d = n & (HDIM - 1); // n = h*HDIM + d
      size_t idx;
      if (mode == 0)
        idx = (((size_t)b_ * NHEADS + h) * SEQ + s) * HDIM + d;
      else
        idx = (((size_t)b_ * NHEADS + h) * HDIM + d) * SEQ + s;
      obf[idx] = (bf16_t)val;
    }
  }
}

__global__ void gemm_xwT(const bf16_t* __restrict__ X,
                         const bf16_t* __restrict__ W,
                         const float* __restrict__ bias,
                         bf16_t* __restrict__ obf,
                         float* __restrict__ of32,
                         int M, int N, int K, int mode, float oscale) {
  int wave = blockIdx.x * (blockDim.x >> 5) + ((int)threadIdx.x >> 5);
  int lane = threadIdx.x & 31;
  int tilesN = N >> 5;
  int totalTiles = (M >> 5) * tilesN;
  if (wave >= totalTiles) return;           // wave-uniform
  int tm = wave / tilesN, tn = wave % tilesN;
  int m0 = tm << 5, n0 = tn << 5;

  v8f acc00 = {}, acc01 = {}, acc10 = {}, acc11 = {};
  for (int k0 = 0; k0 < K; k0 += 32) {
    v16bf a0 = load_a_frag(X, K, m0,      k0, lane);
    v16bf a1 = load_a_frag(X, K, m0 + 16, k0, lane);
    v16bf b0 = load_b_fragT(W, K, n0,      k0, lane);
    v16bf b1 = load_b_fragT(W, K, n0 + 16, k0, lane);
    acc00 = wmma_bf16(a0, b0, acc00);
    acc01 = wmma_bf16(a0, b1, acc01);
    acc10 = wmma_bf16(a1, b0, acc10);
    acc11 = wmma_bf16(a1, b1, acc11);
  }
  store_tile(acc00, m0,      n0,      lane, bias, obf, of32, N, mode, oscale);
  store_tile(acc01, m0,      n0 + 16, lane, bias, obf, of32, N, mode, oscale);
  store_tile(acc10, m0 + 16, n0,      lane, bias, obf, of32, N, mode, oscale);
  store_tile(acc11, m0 + 16, n0 + 16, lane, bias, obf, of32, N, mode, oscale);
}

// ---------------- fused flash attention (block-cooperative) ----------------
// Block = 8 waves = one (b,h) x 128 query rows (16 per wave). K/V tiles for 32
// keys live in double-buffered LDS, filled by the Tensor Data Mover (wave 0)
// and overlapped with compute; all 8 waves share each tile (8x less L2
// traffic than per-wave streaming). Per wave, scores -> online softmax ->
// P relayout (C->A) through a private 1KB LDS bounce -> PV accumulate.
__global__ __launch_bounds__(256) void flash_attn(
    const bf16_t* __restrict__ Qh, const bf16_t* __restrict__ Kh,
    const bf16_t* __restrict__ VhT, bf16_t* __restrict__ ctx) {
  __shared__ __align__(16) bf16_t Ktile[2][32 * HDIM];   // 2 x 4KB
  __shared__ __align__(16) bf16_t Vtile[2][HDIM * 32];   // 2 x 4KB
  __shared__ __align__(16) unsigned char pbuf[8 * 1024]; // 1KB per wave

  int tid = threadIdx.x;
  int wslot = tid >> 5, lane = tid & 31;
  int bh   = blockIdx.x >> 4;               // 64 (b,h) pairs
  int qblk = blockIdx.x & 15;               // 16 query blocks of 128 rows
  int b = bh >> 4, h = bh & (NHEADS - 1);
  int m0 = (qblk << 7) + (wslot << 4);

  const bf16_t* Q  = Qh  + (size_t)bh * SEQ * HDIM;
  const bf16_t* Kg = Kh  + (size_t)bh * SEQ * HDIM;
  const bf16_t* Vg = VhT + (size_t)bh * HDIM * SEQ;

  // Q tile 16x64 as two A-fragments, kept in registers for the whole pass
  v16bf qa0 = load_a_frag(Q, HDIM, m0, 0,  lane);
  v16bf qa1 = load_a_frag(Q, HDIM, m0, 32, lane);

  v8f o0 = {}, o1 = {}, o2 = {}, o3 = {};   // 16 x 64 f32 accumulator
  float rm[8], rl[8];
  #pragma unroll
  for (int j = 0; j < 8; ++j) { rm[j] = -3.0e38f; rl[j] = 0.0f; }

  int r = lane & 15, g = lane >> 4;
  int half = r >> 3, jrow = r & 7;          // P-gather source coords
  unsigned char* wp = pbuf + (wslot << 10);
  v8bf* sp = (v8bf*)wp;
  const bf16_t* lp = (const bf16_t*)wp;

  // prologue: tile 0
  fill_tiles(Ktile[0], Vtile[0], Kg, Vg, 0, tid);
  fill_wait(tid);
  __syncthreads();

  for (int it = 0; it < SEQ / 32; ++it) {   // 64 iterations of 32 keys
    int cur = it & 1;
    if (it + 1 < SEQ / 32)                  // async fill of next tile
      fill_tiles(Ktile[cur ^ 1], Vtile[cur ^ 1], Kg, Vg, (it + 1) << 5, tid);

    const bf16_t* Kl = Ktile[cur];
    const bf16_t* Vl = Vtile[cur];

    // ---- scores: two 16x16 tiles over this key block ----
    v16bf kb00 = load_b_fragT(Kl, HDIM, 0,  0,  lane);
    v16bf kb01 = load_b_fragT(Kl, HDIM, 0,  32, lane);
    v16bf kb10 = load_b_fragT(Kl, HDIM, 16, 0,  lane);
    v16bf kb11 = load_b_fragT(Kl, HDIM, 16, 32, lane);
    v8f s0 = {}, s1 = {};
    s0 = wmma_bf16(qa0, kb00, s0);
    s0 = wmma_bf16(qa1, kb01, s0);
    s1 = wmma_bf16(qa0, kb10, s1);
    s1 = wmma_bf16(qa1, kb11, s1);

    // ---- online softmax over the 32 new keys ----
    float alpha[8];
    #pragma unroll
    for (int j = 0; j < 8; ++j) {
      float t = fmaxf(s0[j], s1[j]);
      t = fmaxf(t, __shfl_xor(t, 1));       // N-reduction; halves stay apart
      t = fmaxf(t, __shfl_xor(t, 2));
      t = fmaxf(t, __shfl_xor(t, 4));
      t = fmaxf(t, __shfl_xor(t, 8));
      float nm = fmaxf(rm[j], t);
      alpha[j] = __expf(rm[j] - nm);
      rm[j] = nm;
      float p0 = __expf(s0[j] - nm);
      float p1 = __expf(s1[j] - nm);
      s0[j] = p0; s1[j] = p1;
      float rs = p0 + p1;
      rs += __shfl_xor(rs, 1);
      rs += __shfl_xor(rs, 2);
      rs += __shfl_xor(rs, 4);
      rs += __shfl_xor(rs, 8);
      rl[j] = rl[j] * alpha[j] + rs;
    }
    #pragma unroll
    for (int j = 0; j < 8; ++j) {
      o0[j] *= alpha[j]; o1[j] *= alpha[j];
      o2[j] *= alpha[j]; o3[j] *= alpha[j];
    }

    // ---- relayout P (C layout f32) -> A fragment (bf16) via private LDS ----
    v8bf pk0, pk1;
    #pragma unroll
    for (int j = 0; j < 8; ++j) { pk0[j] = (bf16_t)s0[j]; pk1[j] = (bf16_t)s1[j]; }
    sp[lane]      = pk0;                    // tile0 packed C layout
    sp[32 + lane] = pk1;                    // tile1 at +512B
    asm volatile("s_wait_dscnt 0" ::: "memory");

    union { v16bf v; bf16_t e[16]; } pa;
    #pragma unroll
    for (int i = 0; i < 8; ++i) {           // v0..v3: K = 8g..8g+7 (tile0)
      int c = 8 * g + i;
      pa.e[i] = lp[((half << 4) + c) * 8 + jrow];
    }
    #pragma unroll
    for (int i = 0; i < 8; ++i) {           // v4..v7: K = 16+8g.. (tile1)
      int c = 8 * g + i;
      pa.e[8 + i] = lp[256 + ((half << 4) + c) * 8 + jrow];
    }

    // ---- O += P @ V_tile (shared LDS V tile, [d][32] layout) ----
    v16bf vb0 = load_b_fragT(Vl, 32, 0,  0, lane);
    v16bf vb1 = load_b_fragT(Vl, 32, 16, 0, lane);
    v16bf vb2 = load_b_fragT(Vl, 32, 32, 0, lane);
    v16bf vb3 = load_b_fragT(Vl, 32, 48, 0, lane);
    o0 = wmma_bf16(pa.v, vb0, o0);
    o1 = wmma_bf16(pa.v, vb1, o1);
    o2 = wmma_bf16(pa.v, vb2, o2);
    o3 = wmma_bf16(pa.v, vb3, o3);

    if (it + 1 < SEQ / 32) fill_wait(tid);  // drain TDM before buffer swap
    __syncthreads();
  }

  // ---- normalize and write merged-head context [B*S][EMBED] ----
  float inv[8];
  #pragma unroll
  for (int j = 0; j < 8; ++j) inv[j] = 1.0f / rl[j];
  int c = lane & 15;
  #pragma unroll
  for (int j = 0; j < 8; ++j) {
    int m = m0 + j + 8 * g;
    size_t base = ((size_t)(b * SEQ + m) * EMBED) + (size_t)h * HDIM + c;
    ctx[base +  0] = (bf16_t)(o0[j] * inv[j]);
    ctx[base + 16] = (bf16_t)(o1[j] * inv[j]);
    ctx[base + 32] = (bf16_t)(o2[j] * inv[j]);
    ctx[base + 48] = (bf16_t)(o3[j] * inv[j]);
  }
}

// ---------------- host-side orchestration ----------------
extern "C" void kernel_launch(void* const* d_in, const int* in_sizes, int n_in,
                              void* d_out, int out_size, void* d_ws,
                              size_t ws_size, hipStream_t stream) {
  (void)in_sizes; (void)n_in; (void)out_size; (void)ws_size;

  const float* query = (const float*)d_in[0];
  const float* key   = (const float*)d_in[1];
  const float* value = (const float*)d_in[2];
  const float* Wq = (const float*)d_in[3];
  const float* bq = (const float*)d_in[4];
  const float* Wk = (const float*)d_in[5];
  const float* bk = (const float*)d_in[6];
  const float* Wv = (const float*)d_in[7];
  const float* bv = (const float*)d_in[8];
  const float* Wo = (const float*)d_in[9];
  const float* bo = (const float*)d_in[10];
  float* out = (float*)d_out;

  char* ws = (char*)d_ws;
  size_t off = 0;
  auto alloc = [&](size_t bytes) -> char* {
    char* p = ws + off;
    off += (bytes + 255) & ~(size_t)255;
    return p;
  };
  bf16_t* xq  = (bf16_t*)alloc((size_t)MTOK * EMBED * 2);
  bf16_t* xk  = (bf16_t*)alloc((size_t)MTOK * EMBED * 2);
  bf16_t* xv  = (bf16_t*)alloc((size_t)MTOK * EMBED * 2);
  bf16_t* wqb = (bf16_t*)alloc((size_t)EMBED * EMBED * 2);
  bf16_t* wkb = (bf16_t*)alloc((size_t)EMBED * EMBED * 2);
  bf16_t* wvb = (bf16_t*)alloc((size_t)EMBED * EMBED * 2);
  bf16_t* wob = (bf16_t*)alloc((size_t)EMBED * EMBED * 2);
  bf16_t* Qh  = (bf16_t*)alloc((size_t)BATCH * NHEADS * SEQ * HDIM * 2);
  bf16_t* Kh  = (bf16_t*)alloc((size_t)BATCH * NHEADS * SEQ * HDIM * 2);
  bf16_t* VhT = (bf16_t*)alloc((size_t)BATCH * NHEADS * HDIM * SEQ * 2);
  bf16_t* ctx = (bf16_t*)alloc((size_t)MTOK * EMBED * 2);

  // 1) convert activations + weights to bf16
  cvt_f32_bf16<<<1024, 256, 0, stream>>>(query, xq, MTOK * EMBED);
  cvt_f32_bf16<<<1024, 256, 0, stream>>>(key,   xk, MTOK * EMBED);
  cvt_f32_bf16<<<1024, 256, 0, stream>>>(value, xv, MTOK * EMBED);
  cvt_f32_bf16<<<512, 256, 0, stream>>>(Wq, wqb, EMBED * EMBED);
  cvt_f32_bf16<<<512, 256, 0, stream>>>(Wk, wkb, EMBED * EMBED);
  cvt_f32_bf16<<<512, 256, 0, stream>>>(Wv, wvb, EMBED * EMBED);
  cvt_f32_bf16<<<512, 256, 0, stream>>>(Wo, wob, EMBED * EMBED);

  // 2) Q/K/V projections (32x32 blocks, 8 waves/block)
  {
    int tiles  = (MTOK / 32) * (EMBED / 32);     // 8192 waves
    int blocks = (tiles + 7) / 8;                // 1024 blocks
    // 1/sqrt(HDIM) folded into Q so flash kernel uses scores directly
    gemm_xwT<<<blocks, 256, 0, stream>>>(xq, wqb, bq, Qh, nullptr,
                                         MTOK, EMBED, EMBED, 0, 0.125f);
    gemm_xwT<<<blocks, 256, 0, stream>>>(xk, wkb, bk, Kh, nullptr,
                                         MTOK, EMBED, EMBED, 0, 1.0f);
    gemm_xwT<<<blocks, 256, 0, stream>>>(xv, wvb, bv, VhT, nullptr,
                                         MTOK, EMBED, EMBED, 1, 1.0f);  // V^T
  }

  // 3) fused attention: 64 (b,h) x 16 query blocks = 1024 blocks
  flash_attn<<<1024, 256, 0, stream>>>(Qh, Kh, VhT, ctx);

  // 4) output projection -> f32 result
  {
    int tiles  = (MTOK / 32) * (EMBED / 32);
    int blocks = (tiles + 7) / 8;
    gemm_xwT<<<blocks, 256, 0, stream>>>(ctx, wob, bo, nullptr, out,
                                         MTOK, EMBED, EMBED, 2, 1.0f);
  }
}
